// Attention_77103252897850
// MI455X (gfx1250) — matrix operand order repaired and checked by reading, not compiled, not run on hardware
//
#include <hip/hip_runtime.h>
#include <stdint.h>

// ---------------------------------------------------------------------------
// Factorized (Tucker) attention for MI455X / gfx1250 (wave32, WMMA f16->f32)
//   B=64, N=256, C=12, H=8, W=8 ; heads (3,2,2)=12, head_dim (4,4,4)=64
//   SCALE = 64**0.25 (reference MULTIPLIES logits by it)
//   K tile staged into LDS by the Tensor Data Mover (padded rows via TDM pad).
// ---------------------------------------------------------------------------

#define BB     64
#define NSEQ   256
#define CC     12
#define HH     8
#define WW     8
#define NHEAD  12
#define DHEAD  64
#define EMB    768                       // C*H*W
#define SCALE_F 2.8284271247461903f      // 64**0.25

typedef __attribute__((ext_vector_type(16))) _Float16 v16h;
typedef __attribute__((ext_vector_type(8)))  _Float16 v8h;
typedef __attribute__((ext_vector_type(8)))  float    v8f;
typedef __attribute__((ext_vector_type(4)))  unsigned v4u;
typedef __attribute__((ext_vector_type(4)))  int      v4i;
typedef __attribute__((ext_vector_type(8)))  int      v8i;

#if __has_builtin(__builtin_amdgcn_tensor_load_to_lds)
#define HAVE_TDM 1
#else
#define HAVE_TDM 0
#endif

static __device__ __forceinline__ v16h pack16(v8h lo, v8h hi) {
    return __builtin_shufflevector(lo, hi, 0,1,2,3,4,5,6,7,8,9,10,11,12,13,14,15);
}
static __device__ __forceinline__ v8f zero8() {
    v8f z = {0.f,0.f,0.f,0.f,0.f,0.f,0.f,0.f};
    return z;
}

// ---------------------------------------------------------------------------
// Kernel A: q/k/v = TLE(x, W0,W1,W2,b) -> head-major f16 [B][12][N][64]
// one block per (b,n); factorized mode products staged through LDS.
// ---------------------------------------------------------------------------
__global__ __launch_bounds__(256) void tle_qkv_kernel(
    const float* __restrict__ x,
    const float* __restrict__ qW0, const float* __restrict__ qW1,
    const float* __restrict__ qW2, const float* __restrict__ qb,
    const float* __restrict__ kW0, const float* __restrict__ kW1,
    const float* __restrict__ kW2, const float* __restrict__ kb,
    const float* __restrict__ vW0, const float* __restrict__ vW1,
    const float* __restrict__ vW2, const float* __restrict__ vb,
    _Float16* __restrict__ qo, _Float16* __restrict__ ko, _Float16* __restrict__ vo)
{
    __shared__ float xs[EMB];
    __shared__ float t1[EMB];
    __shared__ float t2[EMB];

    const int bn = blockIdx.x;           // b*256 + n
    const int b  = bn >> 8;
    const int n  = bn & 255;
    const float* xp = x + (size_t)bn * EMB;

    for (int i = threadIdx.x; i < EMB; i += 256) xs[i] = xp[i];
    __syncthreads();

    const float* W0s[3] = {qW0, kW0, vW0};
    const float* W1s[3] = {qW1, kW1, vW1};
    const float* W2s[3] = {qW2, kW2, vW2};
    const float* Bs [3] = {qb,  kb,  vb };
    _Float16*    Os [3] = {qo,  ko,  vo };

    for (int p = 0; p < 3; ++p) {
        const float* W0 = W0s[p];
        const float* W1 = W1s[p];
        const float* W2 = W2s[p];
        const float* Bi = Bs[p];
        _Float16*    Op = Os[p];

        // stage 1: t1[a,h,w] = sum_c W0[a,c] x[c,h,w]
        for (int i = threadIdx.x; i < EMB; i += 256) {
            const int a = i >> 6, hw = i & 63;
            float s = 0.f;
            #pragma unroll
            for (int c = 0; c < CC; ++c) s += W0[a*CC + c] * xs[c*64 + hw];
            t1[i] = s;
        }
        __syncthreads();
        // stage 2: t2[a,d,w] = sum_h W1[d,h] t1[a,h,w]
        for (int i = threadIdx.x; i < EMB; i += 256) {
            const int a = i >> 6, d = (i >> 3) & 7, w = i & 7;
            float s = 0.f;
            #pragma unroll
            for (int h = 0; h < HH; ++h) s += W1[d*HH + h] * t1[a*64 + h*8 + w];
            t2[i] = s;
        }
        __syncthreads();
        // stage 3: y[a,d,e] = sum_w W2[e,w] t2[a,d,w] + b[a,d,e] -> head scatter
        for (int i = threadIdx.x; i < EMB; i += 256) {
            const int a = i >> 6, d = (i >> 3) & 7, e = i & 7;
            float s = Bi[i];
            #pragma unroll
            for (int w = 0; w < WW; ++w) s += W2[e*WW + w] * t2[a*64 + d*8 + w];
            const int h0 = a >> 2, d0 = a & 3;
            const int h1 = d >> 2, d1 = d & 3;
            const int h2 = e >> 2, d2 = e & 3;
            const int head = h0*4 + h1*2 + h2;
            const int dim  = d0*16 + d1*4 + d2;
            Op[(((size_t)b*NHEAD + head)*NSEQ + n)*DHEAD + dim] = (_Float16)s;
        }
        __syncthreads();
    }
}

// ---------------------------------------------------------------------------
// Kernel B: fused attention per (b,head). 8 waves x 32 query rows.
// K tile (256x64 f16) staged into LDS by TDM with row padding (64 -> 72 halves)
// to break bank conflicts. S = (Q K^T)*SCALE via v_wmma_f32_16x16x32_f16,
// full-row softmax in registers, P transposed C->A layout through per-wave
// LDS, O = P V with V^T staged in LDS.
// ---------------------------------------------------------------------------
__global__ __launch_bounds__(256) void attn_kernel(
    const _Float16* __restrict__ qg, const _Float16* __restrict__ kg,
    const _Float16* __restrict__ vg, _Float16* __restrict__ og)
{
    __shared__ __align__(16) _Float16 Ksh[NSEQ][72];         // TDM-padded rows (36.9 KB)
    __shared__ __align__(16) _Float16 Vt[DHEAD][NSEQ + 8];   // V transposed (33.8 KB)
    __shared__ __align__(16) _Float16 Pbuf[8][16][40];       // per-wave P chunk (10 KB)

    const int tid  = threadIdx.x;
    const int lane = tid & 31;
    const int wave = tid >> 5;
    const int lm   = lane & 15;
    const int g    = lane >> 4;

    const size_t base = (size_t)blockIdx.x * (NSEQ * DHEAD);
    const _Float16* qp = qg + base;
    const _Float16* kp = kg + base;
    const _Float16* vp = vg + base;
    _Float16*       op = og + base;

#if HAVE_TDM
    // ---- Tensor Data Mover: DMA K tile (256 rows x 64 halves) into LDS,
    //      padding +4 DWORDs after every 32 DWORDs -> LDS row stride 72 halves.
    if (wave == 0) {
        const unsigned lds_k = (unsigned)(uintptr_t)&Ksh[0][0];
        const unsigned long long ga = (unsigned long long)(uintptr_t)kp;
        v4u g0;
        g0[0] = 1u;                                            // count=1 (valid D#)
        g0[1] = lds_k;                                         // lds_addr
        g0[2] = (unsigned)(ga & 0xFFFFFFFFu);                  // global_addr[31:0]
        g0[3] = ((unsigned)(ga >> 32) & 0x01FFFFFFu)
              | 0x80000000u;                                   // addr[56:32] | type=2
        v8i g1;
        g1[0] = (1 << 16)      // data_size = 2 bytes
              | (1 << 20)      // pad_enable
              | (4 << 22)      // pad_interval: 32 DWORDs (128 B)
              | (3 << 25);     // pad_amount:   4 DWORDs (16 B)
        g1[1] = (DHEAD & 0xFFFF) << 16;          // tensor_dim0[15:0] = 64
        g1[2] = (NSEQ << 16);                    // dim0 hi = 0 | tensor_dim1 lo = 256
        g1[3] = (DHEAD << 16);                   // dim1 hi = 0 | tile_dim0 = 64
        g1[4] = NSEQ;                            // tile_dim1 = 256, tile_dim2 = 0
        g1[5] = DHEAD;                           // tensor_dim0_stride = 64
        g1[6] = 0;
        g1[7] = 0;
        v4i gz4 = {0, 0, 0, 0};
#if defined(__clang_major__) && (__clang_major__ >= 23)
        v8i gz8 = {0, 0, 0, 0, 0, 0, 0, 0};
        __builtin_amdgcn_tensor_load_to_lds(g0, g1, gz4, gz4, gz8, 0);
#else
        __builtin_amdgcn_tensor_load_to_lds(g0, g1, gz4, gz4, 0);
#endif
        __builtin_amdgcn_s_wait_tensorcnt(0);
    }
#else
    // fallback: cooperative K load into the padded LDS layout
    for (int idx = tid; idx < NSEQ * 8; idx += 256) {
        const int row = idx >> 3, seg = idx & 7;
        *(v8h*)&Ksh[row][seg*8] = *(const v8h*)(kp + row*DHEAD + seg*8);
    }
#endif

    // cooperative load of V, transposed into LDS
    for (int idx = tid; idx < NSEQ * 8; idx += 256) {
        const int row = idx >> 3, seg = idx & 7;
        v8h val = *(const v8h*)(vp + row*DHEAD + seg*8);
        #pragma unroll
        for (int jj = 0; jj < 8; ++jj) Vt[seg*8 + jj][row] = val[jj];
    }
    __syncthreads();

    for (int mt = 0; mt < 2; ++mt) {
        const int qrow = wave*32 + mt*16 + lm;   // this lane's A-matrix row
        // Q A-fragments (16x32 each): lane holds row lm; K = g*8+i / 16+g*8+i
        const v16h qf0 = pack16(*(const v8h*)(qp + qrow*DHEAD +      g*8),
                                *(const v8h*)(qp + qrow*DHEAD + 16 + g*8));
        const v16h qf1 = pack16(*(const v8h*)(qp + qrow*DHEAD + 32 + g*8),
                                *(const v8h*)(qp + qrow*DHEAD + 48 + g*8));

        // ---- S = Q K^T * SCALE, all 16 column tiles kept in registers ----
        v8f sreg[16];
        #pragma unroll
        for (int j = 0; j < 16; ++j) {
            const _Float16* krow = &Ksh[j*16 + lm][0];      // B: lane = column
            const v16h kf0 = pack16(*(const v8h*)(krow +      g*8),
                                    *(const v8h*)(krow + 16 + g*8));
            const v16h kf1 = pack16(*(const v8h*)(krow + 32 + g*8),
                                    *(const v8h*)(krow + 48 + g*8));
            v8f acc = zero8();
            acc = __builtin_amdgcn_wmma_f32_16x16x32_f16(false, qf0, false, kf0,
                                                         (short)0, acc, false, false);
            acc = __builtin_amdgcn_wmma_f32_16x16x32_f16(false, qf1, false, kf1,
                                                         (short)0, acc, false, false);
            sreg[j] = acc * SCALE_F;
        }

        // ---- row softmax (rows live in VGPR index r + 8*g, cols across lanes) ----
        float mrow[8], lrow[8];
        #pragma unroll
        for (int r = 0; r < 8; ++r) mrow[r] = sreg[0][r];
        #pragma unroll
        for (int j = 1; j < 16; ++j)
            #pragma unroll
            for (int r = 0; r < 8; ++r) mrow[r] = fmaxf(mrow[r], sreg[j][r]);
        #pragma unroll
        for (int r = 0; r < 8; ++r) {
            float m = mrow[r];
            m = fmaxf(m, __shfl_xor(m, 1, 32));
            m = fmaxf(m, __shfl_xor(m, 2, 32));
            m = fmaxf(m, __shfl_xor(m, 4, 32));
            m = fmaxf(m, __shfl_xor(m, 8, 32));
            mrow[r] = m;
        }
        #pragma unroll
        for (int r = 0; r < 8; ++r) lrow[r] = 0.f;
        #pragma unroll
        for (int j = 0; j < 16; ++j)
            #pragma unroll
            for (int r = 0; r < 8; ++r) {
                const float e = __expf(sreg[j][r] - mrow[r]);
                sreg[j][r] = e;
                lrow[r] += e;
            }
        #pragma unroll
        for (int r = 0; r < 8; ++r) {
            float s = lrow[r];
            s += __shfl_xor(s, 1, 32);
            s += __shfl_xor(s, 2, 32);
            s += __shfl_xor(s, 4, 32);
            s += __shfl_xor(s, 8, 32);
            lrow[r] = 1.0f / s;
        }

        // ---- O = P V : stage P chunk through LDS (C-layout -> A-layout) ----
        v8f oacc[4] = {zero8(), zero8(), zero8(), zero8()};
        for (int kt = 0; kt < 8; ++kt) {
            #pragma unroll
            for (int jj = 0; jj < 2; ++jj) {
                const int j = kt*2 + jj;
                #pragma unroll
                for (int r = 0; r < 8; ++r)
                    Pbuf[wave][g*8 + r][jj*16 + lm] =
                        (_Float16)(sreg[j][r] * lrow[r]);
            }
            __asm__ __volatile__("s_wait_dscnt 0" ::: "memory");
            const v16h pf = pack16(*(const v8h*)&Pbuf[wave][lm][     g*8],
                                   *(const v8h*)&Pbuf[wave][lm][16 + g*8]);
            #pragma unroll
            for (int nt = 0; nt < 4; ++nt) {
                const v16h vf = pack16(*(const v8h*)&Vt[nt*16 + lm][kt*32 + g*8],
                                       *(const v8h*)&Vt[nt*16 + lm][kt*32 + 16 + g*8]);
                oacc[nt] = __builtin_amdgcn_wmma_f32_16x16x32_f16(false, pf, false, vf,
                                                                  (short)0, oacc[nt],
                                                                  false, false);
            }
            __asm__ __volatile__("" ::: "memory");  // keep next stores after loads
        }

        // ---- store O tile (f16, head-major) ----
        #pragma unroll
        for (int nt = 0; nt < 4; ++nt)
            #pragma unroll
            for (int r = 0; r < 8; ++r)
                op[(wave*32 + mt*16 + g*8 + r)*DHEAD + nt*16 + lm] =
                    (_Float16)oacc[nt][r];
    }
}

// ---------------------------------------------------------------------------
// Kernel C: gather o from head-major layout, final TLE with oW, fp32 output.
// ---------------------------------------------------------------------------
__global__ __launch_bounds__(256) void tle_out_kernel(
    const _Float16* __restrict__ og,
    const float* __restrict__ oW0, const float* __restrict__ oW1,
    const float* __restrict__ oW2, const float* __restrict__ ob,
    float* __restrict__ out)
{
    __shared__ float xs[EMB];
    __shared__ float t1[EMB];
    __shared__ float t2[EMB];

    const int bn = blockIdx.x;
    const int b  = bn >> 8;
    const int n  = bn & 255;

    // gather head-major f16 -> (c,h,w) fp32
    for (int i = threadIdx.x; i < EMB; i += 256) {
        const int c = i >> 6, h = (i >> 3) & 7, w = i & 7;
        const int h0 = c >> 2, d0 = c & 3;
        const int h1 = h >> 2, d1 = h & 3;
        const int h2 = w >> 2, d2 = w & 3;
        const int head = h0*4 + h1*2 + h2;
        const int dim  = d0*16 + d1*4 + d2;
        xs[i] = (float)og[(((size_t)b*NHEAD + head)*NSEQ + n)*DHEAD + dim];
    }
    __syncthreads();

    for (int i = threadIdx.x; i < EMB; i += 256) {
        const int a = i >> 6, hw = i & 63;
        float s = 0.f;
        #pragma unroll
        for (int c = 0; c < CC; ++c) s += oW0[a*CC + c] * xs[c*64 + hw];
        t1[i] = s;
    }
    __syncthreads();
    for (int i = threadIdx.x; i < EMB; i += 256) {
        const int a = i >> 6, d = (i >> 3) & 7, w = i & 7;
        float s = 0.f;
        #pragma unroll
        for (int h = 0; h < HH; ++h) s += oW1[d*HH + h] * t1[a*64 + h*8 + w];
        t2[i] = s;
    }
    __syncthreads();
    for (int i = threadIdx.x; i < EMB; i += 256) {
        const int a = i >> 6, d = (i >> 3) & 7, e = i & 7;
        float s = ob[i];
        #pragma unroll
        for (int w = 0; w < WW; ++w) s += oW2[e*WW + w] * t2[a*64 + d*8 + w];
        out[(size_t)bn * EMB + i] = s;
    }
}

// ---------------------------------------------------------------------------
extern "C" void kernel_launch(void* const* d_in, const int* in_sizes, int n_in,
                              void* d_out, int out_size, void* d_ws, size_t ws_size,
                              hipStream_t stream) {
    (void)in_sizes; (void)n_in; (void)out_size; (void)ws_size;

    const float* x   = (const float*)d_in[0];
    const float* qW0 = (const float*)d_in[1];
    const float* qW1 = (const float*)d_in[2];
    const float* qW2 = (const float*)d_in[3];
    const float* qb  = (const float*)d_in[4];
    const float* kW0 = (const float*)d_in[5];
    const float* kW1 = (const float*)d_in[6];
    const float* kW2 = (const float*)d_in[7];
    const float* kb  = (const float*)d_in[8];
    const float* vW0 = (const float*)d_in[9];
    const float* vW1 = (const float*)d_in[10];
    const float* vW2 = (const float*)d_in[11];
    const float* vb  = (const float*)d_in[12];
    const float* oW0 = (const float*)d_in[13];
    const float* oW1 = (const float*)d_in[14];
    const float* oW2 = (const float*)d_in[15];
    const float* ob  = (const float*)d_in[16];

    const size_t QSZ = (size_t)BB * NHEAD * NSEQ * DHEAD;   // 12.58M halves each
    _Float16* ws  = (_Float16*)d_ws;
    _Float16* qws = ws;
    _Float16* kws = ws + QSZ;
    _Float16* vws = ws + 2*QSZ;
    _Float16* ows = ws + 3*QSZ;

    tle_qkv_kernel<<<dim3(BB * NSEQ), dim3(256), 0, stream>>>(
        x, qW0,qW1,qW2,qb, kW0,kW1,kW2,kb, vW0,vW1,vW2,vb, qws, kws, vws);

    attn_kernel<<<dim3(BB * NHEAD), dim3(256), 0, stream>>>(qws, kws, vws, ows);

    tle_out_kernel<<<dim3(BB * NSEQ), dim3(256), 0, stream>>>(
        ows, oW0, oW1, oW2, ob, (float*)d_out);
}
